// Block_16037407883346
// MI455X (gfx1250) — matrix operand order repaired and checked by reading, not compiled, not run on hardware
//
#include <hip/hip_runtime.h>
#include <math.h>

typedef __attribute__((ext_vector_type(16))) __bf16 v16bf;
typedef __attribute__((ext_vector_type(8)))  __bf16 v8bf;
typedef __attribute__((ext_vector_type(4)))  __bf16 v4bf;
typedef __attribute__((ext_vector_type(8)))  float  v8f;

#define N_EMBD   768
#define N_HEAD   12
#define HEAD_DIM 64
#define BATCH    8
#define TQ       1024
#define TS       257

__device__ __forceinline__ __bf16 f2bf(float v) { return (__bf16)v; }

__device__ __forceinline__ unsigned lds_addr32(const void* p) {
  // generic shared pointer: addr[31:0] is the LDS byte address (ISA aperture rules)
  return (unsigned)(size_t)p;
}

__device__ __forceinline__ void async_copy_b64(unsigned lds_dst, const void* gsrc) {
  asm volatile("global_load_async_to_lds_b64 %0, %1, off"
               :: "v"(lds_dst), "v"(gsrc) : "memory");
}
__device__ __forceinline__ void async_copy_b128(unsigned lds_dst, const void* gsrc) {
  asm volatile("global_load_async_to_lds_b128 %0, %1, off"
               :: "v"(lds_dst), "v"(gsrc) : "memory");
}
__device__ __forceinline__ void wait_async0() {
  asm volatile("s_wait_asynccnt 0x0" ::: "memory");
}

__device__ __forceinline__ v16bf concat8(v8bf lo, v8bf hi) {
  return __builtin_shufflevector(lo, hi, 0, 1, 2, 3, 4, 5, 6, 7,
                                         8, 9, 10, 11, 12, 13, 14, 15);
}

// ---------------- WMMA fragment gathers (CDNA5 wave32 layouts) ----------------
// A operand 16(M)x32(K), row-major tile, stride in elements (multiple of 8).
__device__ __forceinline__ v16bf load_A_frag(const __bf16* a, int stride) {
  const int lane = threadIdx.x & 31;
  const __bf16* p = a + (lane & 15) * stride + (lane >> 4) * 16;
  return concat8(*(const v8bf*)p, *(const v8bf*)(p + 8));
}

// B operand 32(K)x16(N) gathered from an N-major tile t[n][k] (stride mult of 8).
__device__ __forceinline__ v16bf load_Bt_frag(const __bf16* t, int stride) {
  const int lane = threadIdx.x & 31;
  const __bf16* p = t + (lane & 15) * stride + (lane >> 4) * 16;
  return concat8(*(const v8bf*)p, *(const v8bf*)(p + 8));
}

// B operand from the pre-swizzled GEMM tile: lane's 16 values contiguous.
__device__ __forceinline__ v16bf load_Bsw_frag(const __bf16* base) {
  const __bf16* p = base + (threadIdx.x & 31) * 24;
  return concat8(*(const v8bf*)p, *(const v8bf*)(p + 8));
}

// ---------------- LayerNorm (fp32 in -> bf16 out) ----------------
__global__ __launch_bounds__(256)
void ln_bf16_kernel(const float* __restrict__ x, const float* __restrict__ g,
                    const float* __restrict__ b, __bf16* __restrict__ out, int cols) {
  __shared__ float red[256];
  const int row = blockIdx.x, tid = threadIdx.x;
  const float* xr = x + (size_t)row * cols;

  float s = 0.f;
  for (int c = tid; c < cols; c += 256) s += xr[c];
  red[tid] = s; __syncthreads();
  for (int o = 128; o > 0; o >>= 1) { if (tid < o) red[tid] += red[tid + o]; __syncthreads(); }
  const float mean = red[0] / cols;
  __syncthreads();

  float vs = 0.f;
  for (int c = tid; c < cols; c += 256) { float d = xr[c] - mean; vs += d * d; }
  red[tid] = vs; __syncthreads();
  for (int o = 128; o > 0; o >>= 1) { if (tid < o) red[tid] += red[tid + o]; __syncthreads(); }
  const float rstd = rsqrtf(red[0] / cols + 1e-5f);

  for (int c = tid; c < cols; c += 256)
    out[(size_t)row * cols + c] = f2bf((xr[c] - mean) * rstd * g[c] + b[c]);
}

// ---------------- fp32 -> bf16 convert ----------------
__global__ void cvt_f32_bf16_kernel(const float* __restrict__ in, __bf16* __restrict__ out, size_t n) {
  size_t i = (size_t)blockIdx.x * blockDim.x + threadIdx.x;
  if (i < n) out[i] = f2bf(in[i]);
}

// ---------------- WMMA GEMM: C = act(A@B + bias) + residual ----------------
#define BM 64
#define BN 128
#define KT 32
#define ASTR 40  // 80B rows: 8B-aligned async b64 dests, 16B-aligned frag loads

__global__ __launch_bounds__(256)
void gemm_bf16_kernel(const __bf16* __restrict__ A, int lda,
                      const __bf16* __restrict__ Bm, int ldb,
                      const float* __restrict__ bias,
                      const float* __restrict__ residual, int ldr,
                      float* __restrict__ outf, __bf16* __restrict__ outb, int ldc,
                      int M, int N, int K, int gelu) {
  __shared__ __attribute__((aligned(16))) __bf16 As[BM * ASTR];
  __shared__ __attribute__((aligned(16))) __bf16 Bsw[(BN / 16) * 32 * 24];

  const int tid  = threadIdx.x;
  const int lane = tid & 31;
  const int wid  = tid >> 5;
  const int mw   = wid & 3;   // 0..3 -> M offset 16*mw
  const int nw   = wid >> 2;  // 0..1 -> N offset 64*nw
  const int n0   = blockIdx.x * BN;
  const int m0   = blockIdx.y * BM;
  const bool fullA = (m0 + BM) <= M;          // uniform per block
  const unsigned as_base = lds_addr32(&As[0]);

  v8f acc[4] = {};

  for (int k0 = 0; k0 < K; k0 += KT) {
    // stage A: async DMA global->LDS (64B rows) when block fully in-range
    if (fullA) {
#pragma unroll
      for (int it = 0; it < 2; ++it) {
        const int cc = tid + it * 256;        // 512 x 8B chunks
        const int r = cc >> 3, sub = cc & 7;
        async_copy_b64(as_base + (unsigned)(r * 2 * ASTR + sub * 8),
                       &A[(size_t)(m0 + r) * lda + k0 + sub * 4]);
      }
    } else {
      for (int i = tid; i < (BM * KT) / 4; i += 256) {
        const int r = i >> 3;
        const int c = (i & 7) * 4;
        const int gr = m0 + r;
        v4bf val = {};
        if (gr < M) val = *(const v4bf*)&A[(size_t)gr * lda + k0 + c];
        *(v4bf*)&As[r * ASTR + c] = val;
      }
    }
    // stage B: coalesced b64 global loads -> swizzled u16 scatter so that each
    // (16-col tile, lane) owns 16 contiguous K-values for b128 fragment loads.
    for (int i = tid; i < (KT * BN) / 4; i += 256) {
      const int r  = i >> 5;
      const int c4 = (i & 31) * 4;
      const v4bf val = *(const v4bf*)&Bm[(size_t)(k0 + r) * ldb + n0 + c4];
      const int kh = r >> 4, j = r & 15;
#pragma unroll
      for (int e = 0; e < 4; ++e) {
        const int c = c4 + e;
        Bsw[(((c >> 4) * 32) + kh * 16 + (c & 15)) * 24 + j] = val[e];
      }
    }
    if (fullA) wait_async0();
    __syncthreads();

    if (k0 + KT < K)  // hint next B tile into cache: global_prefetch_b8
      __builtin_prefetch(&Bm[(size_t)(k0 + KT) * ldb + n0 + (size_t)lane * 4], 0, 0);

    // gather all operands first (one DS clause, single dscnt wait),
    // then issue the four WMMAs back-to-back
    const v16bf af = load_A_frag(&As[(16 * mw) * ASTR], ASTR);
    v16bf bfr[4];
#pragma unroll
    for (int nt = 0; nt < 4; ++nt)
      bfr[nt] = load_Bsw_frag(&Bsw[(nw * 4 + nt) * 32 * 24]);
#pragma unroll
    for (int nt = 0; nt < 4; ++nt)
      acc[nt] = __builtin_amdgcn_wmma_f32_16x16x32_bf16(false, af, false, bfr[nt],
                                                        (short)0, acc[nt], false, false);
    __syncthreads();
  }

  const int col = lane & 15, kh = lane >> 4;
#pragma unroll
  for (int nt = 0; nt < 4; ++nt) {
    const int gc = n0 + nw * 64 + nt * 16 + col;
    const float bv = bias ? bias[gc] : 0.f;
#pragma unroll
    for (int r = 0; r < 8; ++r) {
      const int gr = m0 + mw * 16 + kh * 8 + r;
      if (gr < M) {
        float v = acc[nt][r] + bv;
        if (gelu) {
          const float u = 0.7978845608028654f * (v + 0.044715f * v * v * v);
          v = 0.5f * v * (1.0f + tanhf(u));
        }
        if (residual) v += residual[(size_t)gr * ldr + gc];
        if (outf) outf[(size_t)gr * ldc + gc] = v;
        if (outb) outb[(size_t)gr * ldc + gc] = f2bf(v);
      }
    }
  }
}

// ---------------- Flash attention (one wave, 16 query rows, 32-key chunks) ----
// Ks stays [key][dim] (contiguous K^T gathers); V staged transposed [dim][key].
template <int CAUSAL>
__global__ __launch_bounds__(32)
void attn_kernel(const __bf16* __restrict__ Qp, int q_rs,
                 const __bf16* __restrict__ Kp, int k_rs,
                 const __bf16* __restrict__ Vp, int v_rs,
                 __bf16* __restrict__ Op, int o_rs,
                 int Tq, int Tk, float scale) {
  __shared__ __attribute__((aligned(16))) __bf16 Qs[16 * 72];
  __shared__ __attribute__((aligned(16))) __bf16 Ks[32 * 72];
  __shared__ __attribute__((aligned(16))) __bf16 Vt[64 * 40];
  __shared__ __attribute__((aligned(16))) __bf16 Ps[16 * 40];

  const int tid = threadIdx.x;
  const int q0  = blockIdx.x * 16;
  const int h   = blockIdx.y;
  const int bz  = blockIdx.z;

  const __bf16* qb = Qp + (size_t)bz * Tq * q_rs + h * HEAD_DIM;
  const __bf16* kb = Kp + (size_t)bz * Tk * k_rs + h * HEAD_DIM;
  const __bf16* vb = Vp + (size_t)bz * Tk * v_rs + h * HEAD_DIM;
  __bf16*       ob = Op + (size_t)bz * Tq * o_rs + h * HEAD_DIM;

  const unsigned qs_base = lds_addr32(&Qs[0]);
  const unsigned ks_base = lds_addr32(&Ks[0]);

  // stage Q via async DMA: 16 rows x 128B -> 128 x 16B chunks
#pragma unroll
  for (int it = 0; it < 4; ++it) {
    const int cc = tid + it * 32;
    const int r = cc >> 3, sub = cc & 7;
    async_copy_b128(qs_base + (unsigned)(r * 144 + sub * 16),
                    &qb[(size_t)(q0 + r) * q_rs + sub * 8]);
  }
  wait_async0();
  __syncthreads();
  const v16bf aq0 = load_A_frag(&Qs[0], 72);   // dims 0..31
  const v16bf aq1 = load_A_frag(&Qs[32], 72);  // dims 32..63

  const int col = tid & 15, kh = tid >> 4;
  float mrow[8], lrow[8];
  v8f oacc[4] = {};
#pragma unroll
  for (int r = 0; r < 8; ++r) { mrow[r] = -3.0e38f; lrow[r] = 0.f; }

  const int kend = CAUSAL ? (q0 + 16) : Tk;
  for (int kc = 0; kc < kend; kc += 32) {
    const bool fullk = (kc + 32) <= Tk;        // uniform
    if (fullk) {
      // K tile: async DMA, 32 rows x 128B -> 256 x 16B chunks
#pragma unroll
      for (int it = 0; it < 8; ++it) {
        const int cc = tid + it * 32;
        const int r = cc >> 3, sub = cc & 7;
        async_copy_b128(ks_base + (unsigned)(r * 144 + sub * 16),
                        &kb[(size_t)(kc + r) * k_rs + sub * 8]);
      }
    } else {
      for (int i = tid; i < 32 * 16; i += 32) {
        const int r = i >> 4, c = (i & 15) * 4;
        const int gk = kc + r;
        v4bf kv = {};
        if (gk < Tk) kv = *(const v4bf*)&kb[(size_t)gk * k_rs + c];
        *(v4bf*)&Ks[r * 72 + c] = kv;
      }
    }
    // V tile: VGPR-mediated transpose scatter into Vt[dim][key]
    for (int i = tid; i < 32 * 16; i += 32) {
      const int r = i >> 4, c = (i & 15) * 4;
      const int gk = kc + r;
      v4bf vv = {};
      if (gk < Tk) vv = *(const v4bf*)&vb[(size_t)gk * v_rs + c];
#pragma unroll
      for (int e = 0; e < 4; ++e) Vt[(c + e) * 40 + r] = vv[e];
    }
    if (fullk) wait_async0();
    __syncthreads();

    // S = Q @ K^T : gather all four B fragments, then chain the WMMAs
    v8f s0 = {}, s1 = {};
    {
      const v16bf b00 = load_Bt_frag(&Ks[0], 72);
      const v16bf b01 = load_Bt_frag(&Ks[32], 72);
      const v16bf b10 = load_Bt_frag(&Ks[16 * 72], 72);
      const v16bf b11 = load_Bt_frag(&Ks[16 * 72 + 32], 72);
      s0 = __builtin_amdgcn_wmma_f32_16x16x32_bf16(false, aq0, false, b00, (short)0, s0, false, false);
      s0 = __builtin_amdgcn_wmma_f32_16x16x32_bf16(false, aq1, false, b01, (short)0, s0, false, false);
      s1 = __builtin_amdgcn_wmma_f32_16x16x32_bf16(false, aq0, false, b10, (short)0, s1, false, false);
      s1 = __builtin_amdgcn_wmma_f32_16x16x32_bf16(false, aq1, false, b11, (short)0, s1, false, false);
    }

    // online softmax over this chunk (row stats per lane for rows kh*8+r)
    float p0[8], p1[8], cf[8];
#pragma unroll
    for (int r = 0; r < 8; ++r) {
      const int qi  = q0 + kh * 8 + r;
      const int k0i = kc + col;
      const int k1i = kc + 16 + col;
      float v0 = s0[r] * scale, v1 = s1[r] * scale;
      if (k0i >= Tk || (CAUSAL && k0i > qi)) v0 = -3.0e38f;
      if (k1i >= Tk || (CAUSAL && k1i > qi)) v1 = -3.0e38f;
      float mx = fmaxf(v0, v1);
      for (int o = 8; o >= 1; o >>= 1) mx = fmaxf(mx, __shfl_xor(mx, o, 16));
      const float mn = fmaxf(mrow[r], mx);
      const float c  = __expf(mrow[r] - mn);
      const float e0 = __expf(v0 - mn);
      const float e1 = __expf(v1 - mn);
      float ps = e0 + e1;
      for (int o = 8; o >= 1; o >>= 1) ps += __shfl_xor(ps, o, 16);
      lrow[r] = lrow[r] * c + ps;
      mrow[r] = mn;
      cf[r] = c; p0[r] = e0; p1[r] = e1;
    }

    // write P (C-frag order) into LDS in A-tile layout; rescale O
#pragma unroll
    for (int r = 0; r < 8; ++r) {
      Ps[(kh * 8 + r) * 40 + col]      = f2bf(p0[r]);
      Ps[(kh * 8 + r) * 40 + 16 + col] = f2bf(p1[r]);
#pragma unroll
      for (int nt = 0; nt < 4; ++nt) oacc[nt][r] *= cf[r];
    }
    __syncthreads();

    // O += P @ V : gather all operands first, then chain the WMMAs
    const v16bf ap = load_A_frag(&Ps[0], 40);
    v16bf bv[4];
#pragma unroll
    for (int nt = 0; nt < 4; ++nt)
      bv[nt] = load_Bt_frag(&Vt[nt * 16 * 40], 40);
#pragma unroll
    for (int nt = 0; nt < 4; ++nt)
      oacc[nt] = __builtin_amdgcn_wmma_f32_16x16x32_bf16(false, ap, false, bv[nt],
                                                         (short)0, oacc[nt], false, false);
    __syncthreads();
  }

#pragma unroll
  for (int nt = 0; nt < 4; ++nt) {
#pragma unroll
    for (int r = 0; r < 8; ++r) {
      const int qr = q0 + kh * 8 + r;
      ob[(size_t)qr * o_rs + nt * 16 + col] = f2bf(oacc[nt][r] / lrow[r]);
    }
  }
}

// ---------------- host orchestration ----------------
extern "C" void kernel_launch(void* const* d_in, const int* in_sizes, int n_in,
                              void* d_out, int out_size, void* d_ws, size_t ws_size,
                              hipStream_t stream) {
  (void)in_sizes; (void)n_in; (void)out_size; (void)ws_size;
  const float* x            = (const float*)d_in[0];
  const float* enc          = (const float*)d_in[1];
  const float* ln1_g        = (const float*)d_in[2];
  const float* ln1_b        = (const float*)d_in[3];
  const float* ln2_g        = (const float*)d_in[4];
  const float* ln2_b        = (const float*)d_in[5];
  const float* ln3_g        = (const float*)d_in[6];
  const float* ln3_b        = (const float*)d_in[7];
  const float* attn_w       = (const float*)d_in[8];
  const float* attn_b       = (const float*)d_in[9];
  const float* attn_proj_w  = (const float*)d_in[10];
  const float* attn_proj_b  = (const float*)d_in[11];
  const float* cross_w      = (const float*)d_in[12];
  const float* cross_b      = (const float*)d_in[13];
  const float* cross_proj_w = (const float*)d_in[14];
  const float* cross_proj_b = (const float*)d_in[15];
  const float* fc_w         = (const float*)d_in[16];
  const float* fc_b         = (const float*)d_in[17];
  const float* proj_w       = (const float*)d_in[18];
  const float* proj_b       = (const float*)d_in[19];
  float* out = (float*)d_out;

  const size_t BT = (size_t)BATCH * TQ;  // 8192
  const size_t BS = (size_t)BATCH * TS;  // 2056

  char* ws = (char*)d_ws;
  size_t off = 0;
  auto take = [&](size_t bytes) -> char* {
    char* p = ws + off;
    off += (bytes + 255) & ~(size_t)255;
    return p;
  };
  __bf16* hbuf  = (__bf16*)take(BT * N_EMBD * 2);          // LN outputs (reused)
  __bf16* kvbuf = (__bf16*)take(BT * 3 * N_EMBD * 2);      // qkv / (qc + ekv)
  __bf16* obuf  = (__bf16*)take(BT * N_EMBD * 2);          // attn out / enc bf16
  float*  xa    = (float*)take(BT * N_EMBD * 4);           // x after self-attn
  float*  xb    = (float*)take(BT * N_EMBD * 4);           // x after cross-attn
  __bf16* wbuf  = (__bf16*)take((size_t)N_EMBD * 4 * N_EMBD * 2);  // weight scratch
  __bf16* fcb   = (__bf16*)take(BT * 4 * N_EMBD * 2);      // MLP hidden
  __bf16* ebuf  = obuf;                                    // encoder bf16 (reuse)
  __bf16* qc    = kvbuf;                                   // cross Q [BT,768]
  __bf16* ekv   = kvbuf + BT * N_EMBD;                     // cross K,V [BS,1536]

  const dim3 b256(256);
  const float scale = 0.125f;  // 1/sqrt(64)
  auto cvt = [&](const float* src, __bf16* dst, size_t n) {
    cvt_f32_bf16_kernel<<<(unsigned)((n + 255) / 256), b256, 0, stream>>>(src, dst, n);
  };

  // ---- causal self-attention ----
  ln_bf16_kernel<<<(unsigned)BT, b256, 0, stream>>>(x, ln1_g, ln1_b, hbuf, N_EMBD);
  cvt(attn_w, wbuf, (size_t)N_EMBD * 3 * N_EMBD);
  gemm_bf16_kernel<<<dim3(3 * N_EMBD / BN, (unsigned)((BT + BM - 1) / BM)), b256, 0, stream>>>(
      hbuf, N_EMBD, wbuf, 3 * N_EMBD, attn_b, nullptr, 0, nullptr, kvbuf, 3 * N_EMBD,
      (int)BT, 3 * N_EMBD, N_EMBD, 0);
  attn_kernel<1><<<dim3(TQ / 16, N_HEAD, BATCH), 32, 0, stream>>>(
      kvbuf, 3 * N_EMBD, kvbuf + N_EMBD, 3 * N_EMBD, kvbuf + 2 * N_EMBD, 3 * N_EMBD,
      obuf, N_EMBD, TQ, TQ, scale);
  cvt(attn_proj_w, wbuf, (size_t)N_EMBD * N_EMBD);
  gemm_bf16_kernel<<<dim3(N_EMBD / BN, (unsigned)((BT + BM - 1) / BM)), b256, 0, stream>>>(
      obuf, N_EMBD, wbuf, N_EMBD, attn_proj_b, x, N_EMBD, xa, nullptr, N_EMBD,
      (int)BT, N_EMBD, N_EMBD, 0);

  // ---- cross-attention ----
  ln_bf16_kernel<<<(unsigned)BT, b256, 0, stream>>>(xa, ln2_g, ln2_b, hbuf, N_EMBD);
  cvt(cross_w, wbuf, (size_t)N_EMBD * 3 * N_EMBD);
  gemm_bf16_kernel<<<dim3(N_EMBD / BN, (unsigned)((BT + BM - 1) / BM)), b256, 0, stream>>>(
      hbuf, N_EMBD, wbuf, 3 * N_EMBD, cross_b, nullptr, 0, nullptr, qc, N_EMBD,
      (int)BT, N_EMBD, N_EMBD, 0);
  cvt(enc, ebuf, BS * N_EMBD);
  gemm_bf16_kernel<<<dim3(2 * N_EMBD / BN, (unsigned)((BS + BM - 1) / BM)), b256, 0, stream>>>(
      ebuf, N_EMBD, wbuf + N_EMBD, 3 * N_EMBD, cross_b + N_EMBD, nullptr, 0,
      nullptr, ekv, 2 * N_EMBD, (int)BS, 2 * N_EMBD, N_EMBD, 0);
  attn_kernel<0><<<dim3(TQ / 16, N_HEAD, BATCH), 32, 0, stream>>>(
      qc, N_EMBD, ekv, 2 * N_EMBD, ekv + N_EMBD, 2 * N_EMBD,
      obuf, N_EMBD, TQ, TS, scale);
  cvt(cross_proj_w, wbuf, (size_t)N_EMBD * N_EMBD);
  gemm_bf16_kernel<<<dim3(N_EMBD / BN, (unsigned)((BT + BM - 1) / BM)), b256, 0, stream>>>(
      obuf, N_EMBD, wbuf, N_EMBD, cross_proj_b, xa, N_EMBD, xb, nullptr, N_EMBD,
      (int)BT, N_EMBD, N_EMBD, 0);

  // ---- MLP ----
  ln_bf16_kernel<<<(unsigned)BT, b256, 0, stream>>>(xb, ln3_g, ln3_b, hbuf, N_EMBD);
  cvt(fc_w, wbuf, (size_t)N_EMBD * 4 * N_EMBD);
  gemm_bf16_kernel<<<dim3(4 * N_EMBD / BN, (unsigned)((BT + BM - 1) / BM)), b256, 0, stream>>>(
      hbuf, N_EMBD, wbuf, 4 * N_EMBD, fc_b, nullptr, 0, nullptr, fcb, 4 * N_EMBD,
      (int)BT, 4 * N_EMBD, N_EMBD, 1);
  cvt(proj_w, wbuf, (size_t)4 * N_EMBD * N_EMBD);
  gemm_bf16_kernel<<<dim3(N_EMBD / BN, (unsigned)((BT + BM - 1) / BM)), b256, 0, stream>>>(
      fcb, 4 * N_EMBD, wbuf, N_EMBD, proj_b, xb, N_EMBD, out, nullptr, N_EMBD,
      (int)BT, N_EMBD, 4 * N_EMBD, 0);
}
